// MHA_15934328668243
// MI455X (gfx1250) — matrix operand order repaired and checked by reading, not compiled, not run on hardware
//
#include <hip/hip_runtime.h>
#include <hip/hip_bf16.h>

// ---------------- problem constants (hardcoded, all divide evenly) ---------
#define EDIM   1024
#define NHEADS 16
#define DHEAD  64
#define BATCH  4
#define SEQ    2048
#define TOKENS 8192   // BATCH*SEQ

typedef __attribute__((ext_vector_type(16))) __bf16 v16bf;
typedef __attribute__((ext_vector_type(8)))  float  v8f;
typedef __attribute__((ext_vector_type(4)))  int    v4i;

union Frag {
    uint4 q[2];
    v16bf v;
};

__device__ __forceinline__ unsigned short f32_to_bf16(float f) {
    unsigned u = __float_as_uint(f);
    unsigned r = u + 0x7FFFu + ((u >> 16) & 1u);   // round-to-nearest-even
    return (unsigned short)(r >> 16);
}

// ---- CDNA5 async global->LDS copy (ASYNCcnt path), with safe fallback -----
#if __has_builtin(__builtin_amdgcn_global_load_async_to_lds_b128)
#define HAVE_ASYNC_LDS 1
#else
#define HAVE_ASYNC_LDS 0
#endif

typedef __attribute__((address_space(1))) v4i* gptr_b128;   // global int4*
typedef __attribute__((address_space(3))) v4i* lptr_b128;   // LDS    int4*

__device__ __forceinline__ void async_copy_b128(const unsigned short* g, unsigned short* l) {
#if HAVE_ASYNC_LDS
    v4i* gg = (v4i*)const_cast<unsigned short*>(g);
    v4i* ll = (v4i*)l;
    __builtin_amdgcn_global_load_async_to_lds_b128((gptr_b128)gg, (lptr_b128)ll, 0, 0);
#else
    *(uint4*)l = *(const uint4*)g;
#endif
}

__device__ __forceinline__ void wait_async_copies() {
#if HAVE_ASYNC_LDS
#if __has_builtin(__builtin_amdgcn_s_wait_asynccnt)
    __builtin_amdgcn_s_wait_asynccnt(0);
#else
    asm volatile("s_wait_asynccnt 0x0" ::: "memory");
#endif
#endif
}

// ---------------------------------------------------------------------------
// Weight transpose + f32->bf16 convert:  Wt[n*E+k] = bf16(W[k*E+n])
// ---------------------------------------------------------------------------
__global__ __launch_bounds__(256) void conv_transpose_w(
    const float* __restrict__ W, unsigned short* __restrict__ Wt) {
    __shared__ float tile[32][33];
    const int bx = blockIdx.x * 32;      // n-tile
    const int by = blockIdx.y * 32;      // k-tile
    const int tx = threadIdx.x & 31;
    const int ty = threadIdx.x >> 5;     // 0..7
#pragma unroll
    for (int i = 0; i < 32; i += 8)
        tile[ty + i][tx] = W[(long)(by + ty + i) * EDIM + bx + tx];
    __syncthreads();
#pragma unroll
    for (int i = 0; i < 32; i += 8)
        Wt[(long)(bx + ty + i) * EDIM + by + tx] = f32_to_bf16(tile[tx][ty + i]);
}

// ---------------------------------------------------------------------------
// Generic bf16 WMMA GEMM with double-buffered LDS + async global->LDS fills.
//   A: f32 (converted on stage) if !A_BF16, else bf16 row-major [m][k]
//   Bt: bf16 [n][k] row-major
//   MODE 0: C f32 row-major      MODE 1: C bf16 row-major
//   MODE 2: C bf16 "V-transpose": (m=token,n=embed) -> [(b*H+h)*Dh+d]*SEQ + t
// Batched over blockIdx.z with (z/H, z%H) stride pairs.
// ---------------------------------------------------------------------------
template<int BM, int BN, int WM, int WN, int MODE, bool HAS_BIAS, bool USE_SCALE, bool A_BF16>
__global__ __launch_bounds__(256) void gemm_bf16(
    const void* __restrict__ Asrc, const unsigned short* __restrict__ Bt,
    const float* __restrict__ bias, float scale, void* __restrict__ C,
    int K, int lda, int ldb, int ldc,
    long ASb, long ASh, long BSb, long BSh, long CSb, long CSh) {

    constexpr int KC   = 32;
    constexpr int AROW = KC + 8;               // ushort units, 80B rows (16B aligned)
    constexpr int ITM  = BM / (WM * 16);
    constexpr int ITN  = BN / (WN * 16);

    __shared__ unsigned short As[2][BM * AROW];
    __shared__ unsigned short Bs[2][BN * AROW];

    const int tid  = threadIdx.x;
    const int wave = tid >> 5;
    const int lane = tid & 31;
    const int l16  = lane & 15;
    const int lhi  = lane >> 4;                // 0 / 1

    const int z   = blockIdx.z;
    const long zA = (long)(z / NHEADS) * ASb + (long)(z % NHEADS) * ASh;
    const long zB = (long)(z / NHEADS) * BSb + (long)(z % NHEADS) * BSh;
    const long zC = (long)(z / NHEADS) * CSb + (long)(z % NHEADS) * CSh;

    const int bm0 = blockIdx.x * BM;
    const int bn0 = blockIdx.y * BN;
    const int wm  = wave % WM;
    const int wn  = wave / WM;
    const int wRow = bm0 + wm * ITM * 16;
    const int wCol = bn0 + wn * ITN * 16;

    // stage one KC-deep slab of A and B into LDS buffer `buf`
    auto stage = [&](int buf, int k0) {
        const int r  = tid >> 2;               // 0..63
        const int kc = (tid & 3) * 8;          // 0,8,16,24
        if constexpr (A_BF16) {
            const unsigned short* Ab = (const unsigned short*)Asrc;
#pragma unroll
            for (int it = 0; it < BM / 64; ++it) {
                const int row = r + it * 64;
                async_copy_b128(Ab + zA + (long)(bm0 + row) * lda + k0 + kc,
                                &As[buf][row * AROW + kc]);
            }
        } else {
            const float* Af = (const float*)Asrc;
#pragma unroll
            for (int it = 0; it < BM / 64; ++it) {
                const int row = r + it * 64;
                const float* src = Af + zA + (long)(bm0 + row) * lda + k0 + kc;
                const float4 f0 = *(const float4*)(src);
                const float4 f1 = *(const float4*)(src + 4);
                uint4 p;
                p.x = (unsigned)f32_to_bf16(f0.x) | ((unsigned)f32_to_bf16(f0.y) << 16);
                p.y = (unsigned)f32_to_bf16(f0.z) | ((unsigned)f32_to_bf16(f0.w) << 16);
                p.z = (unsigned)f32_to_bf16(f1.x) | ((unsigned)f32_to_bf16(f1.y) << 16);
                p.w = (unsigned)f32_to_bf16(f1.z) | ((unsigned)f32_to_bf16(f1.w) << 16);
                *(uint4*)&As[buf][row * AROW + kc] = p;
            }
        }
#pragma unroll
        for (int it = 0; it < BN / 64; ++it) {
            const int row = r + it * 64;
            async_copy_b128(Bt + zB + (long)(bn0 + row) * ldb + k0 + kc,
                            &Bs[buf][row * AROW + kc]);
        }
    };

    v8f acc[ITM][ITN] = {};

    stage(0, 0);
    wait_async_copies();
    __syncthreads();

    const int nk = K / KC;
    for (int ik = 0; ik < nk; ++ik) {
        const int buf = ik & 1;
        if (ik + 1 < nk) stage(buf ^ 1, (ik + 1) * KC);   // overlap with WMMA below

        // ---- load fragments per ISA 7.12.2 layouts
        Frag a[ITM], b[ITN];
#pragma unroll
        for (int i = 0; i < ITM; ++i) {
            const unsigned short* src = &As[buf][(wm * ITM * 16 + i * 16 + l16) * AROW];
            a[i].q[0] = *(const uint4*)(src + lhi * 8);        // K 0-7  / 8-15
            a[i].q[1] = *(const uint4*)(src + 16 + lhi * 8);   // K 16-23 / 24-31
        }
#pragma unroll
        for (int j = 0; j < ITN; ++j) {
            const unsigned short* src = &Bs[buf][(wn * ITN * 16 + j * 16 + l16) * AROW];
            b[j].q[0] = *(const uint4*)(src + lhi * 8);        // K 0-7  / 8-15
            b[j].q[1] = *(const uint4*)(src + 16 + lhi * 8);   // K 16-23 / 24-31
        }

#pragma unroll
        for (int i = 0; i < ITM; ++i)
#pragma unroll
            for (int j = 0; j < ITN; ++j)
                acc[i][j] = __builtin_amdgcn_wmma_f32_16x16x32_bf16(
                    false, a[i].v, false, b[j].v, (short)0, acc[i][j], false, false);

        wait_async_copies();   // next buffer's async fills done (per-wave)
        __syncthreads();       // all waves done reading `buf` and done staging
    }

    // ---- store D (VGPR r: row = r + 8*lhi, col = l16)
#pragma unroll
    for (int j = 0; j < ITN; ++j) {
        const int col = wCol + j * 16 + l16;
        const float bv = HAS_BIAS ? bias[col] : 0.0f;
#pragma unroll
        for (int i = 0; i < ITM; ++i) {
#pragma unroll
            for (int r = 0; r < 8; ++r) {
                const int row = wRow + i * 16 + r + lhi * 8;
                float val = acc[i][j][r];
                if (USE_SCALE) val *= scale;
                if (HAS_BIAS)  val += bv;
                if constexpr (MODE == 0) {
                    ((float*)C)[zC + (long)row * ldc + col] = val;
                } else if constexpr (MODE == 1) {
                    ((unsigned short*)C)[zC + (long)row * ldc + col] = f32_to_bf16(val);
                } else {
                    // V transpose: b=row/SEQ, t=row%SEQ, h=col/DHEAD, d=col%DHEAD
                    const long idx = ((long)(row >> 11) * NHEADS + (col >> 6)) * ((long)DHEAD * SEQ)
                                   + (long)(col & 63) * SEQ + (row & 2047);
                    ((unsigned short*)C)[idx] = f32_to_bf16(val);
                }
            }
        }
    }
}

// ---------------------------------------------------------------------------
// In-place row softmax over attn rows of length SEQ (one block per row)
// ---------------------------------------------------------------------------
__global__ __launch_bounds__(256) void softmax_rows_k(float* __restrict__ attn) {
    __shared__ float red[256];
    const long row = blockIdx.x;
    float* p = attn + row * (long)SEQ;
    const int tid = threadIdx.x;

    float4 f0 = *(float4*)(p + tid * 8);
    float4 f1 = *(float4*)(p + tid * 8 + 4);
    float x[8] = {f0.x, f0.y, f0.z, f0.w, f1.x, f1.y, f1.z, f1.w};

    float m = x[0];
#pragma unroll
    for (int i = 1; i < 8; ++i) m = fmaxf(m, x[i]);
    red[tid] = m;
    __syncthreads();
    for (int s = 128; s > 0; s >>= 1) {
        if (tid < s) red[tid] = fmaxf(red[tid], red[tid + s]);
        __syncthreads();
    }
    m = red[0];
    __syncthreads();

    float sum = 0.0f;
#pragma unroll
    for (int i = 0; i < 8; ++i) { x[i] = __expf(x[i] - m); sum += x[i]; }
    red[tid] = sum;
    __syncthreads();
    for (int s = 128; s > 0; s >>= 1) {
        if (tid < s) red[tid] += red[tid + s];
        __syncthreads();
    }
    const float inv = 1.0f / red[0];

    *(float4*)(p + tid * 8)     = make_float4(x[0]*inv, x[1]*inv, x[2]*inv, x[3]*inv);
    *(float4*)(p + tid * 8 + 4) = make_float4(x[4]*inv, x[5]*inv, x[6]*inv, x[7]*inv);
}

// ---------------------------------------------------------------------------
extern "C" void kernel_launch(void* const* d_in, const int* in_sizes, int n_in,
                              void* d_out, int out_size, void* d_ws, size_t ws_size,
                              hipStream_t stream) {
    (void)in_sizes; (void)n_in; (void)out_size; (void)ws_size;

    const float* query = (const float*)d_in[0];
    const float* key   = (const float*)d_in[1];
    const float* value = (const float*)d_in[2];
    const float* Wq    = (const float*)d_in[3];
    const float* bq    = (const float*)d_in[4];
    const float* Wk    = (const float*)d_in[5];
    const float* bk    = (const float*)d_in[6];
    const float* Wv    = (const float*)d_in[7];
    const float* bv    = (const float*)d_in[8];
    const float* Wo    = (const float*)d_in[9];
    const float* bo    = (const float*)d_in[10];

    // d_out = [ out (TOKENS*EDIM f32) | attn (B*H*S*S f32) ]
    float* out  = (float*)d_out;
    float* attn = (float*)d_out + (long)TOKENS * EDIM;

    // workspace carve-up (72 MiB total)
    char* ws = (char*)d_ws;
    unsigned short* Wq_t = (unsigned short*)(ws + (0ull  << 20));
    unsigned short* Wk_t = (unsigned short*)(ws + (2ull  << 20));
    unsigned short* Wv_t = (unsigned short*)(ws + (4ull  << 20));
    unsigned short* Wo_t = (unsigned short*)(ws + (6ull  << 20));
    unsigned short* Qp   = (unsigned short*)(ws + (8ull  << 20));  // [TOKENS][E] bf16
    unsigned short* Kp   = (unsigned short*)(ws + (24ull << 20));  // [TOKENS][E] bf16
    unsigned short* Vt   = (unsigned short*)(ws + (40ull << 20));  // [B][H][Dh][S] bf16
    unsigned short* ctx  = (unsigned short*)(ws + (56ull << 20));  // [TOKENS][E] bf16

    const dim3 tblk(256);

    // 1) weight transpose + bf16 convert
    {
        dim3 g(EDIM / 32, EDIM / 32);
        conv_transpose_w<<<g, tblk, 0, stream>>>(Wq, Wq_t);
        conv_transpose_w<<<g, tblk, 0, stream>>>(Wk, Wk_t);
        conv_transpose_w<<<g, tblk, 0, stream>>>(Wv, Wv_t);
        conv_transpose_w<<<g, tblk, 0, stream>>>(Wo, Wo_t);
    }

    // 2) projections: [TOKENS,E] x [E,E] + b  (A = f32 inputs, convert on stage)
    {
        dim3 g(TOKENS / 128, EDIM / 128, 1);
        gemm_bf16<128,128,2,4,1,true,false,false><<<g, tblk, 0, stream>>>(
            query, Wq_t, bq, 1.0f, (void*)Qp, EDIM, EDIM, EDIM, EDIM,
            0,0, 0,0, 0,0);
        gemm_bf16<128,128,2,4,1,true,false,false><<<g, tblk, 0, stream>>>(
            key,   Wk_t, bk, 1.0f, (void*)Kp, EDIM, EDIM, EDIM, EDIM,
            0,0, 0,0, 0,0);
        gemm_bf16<128,128,2,4,2,true,false,false><<<g, tblk, 0, stream>>>(
            value, Wv_t, bv, 1.0f, (void*)Vt, EDIM, EDIM, EDIM, EDIM,
            0,0, 0,0, 0,0);
    }

    // 3) scores = (Q Kt) / sqrt(Dh), per (b,h): M=N=SEQ, K=Dh  (A bf16 -> async)
    {
        dim3 g(SEQ / 128, SEQ / 128, BATCH * NHEADS);
        gemm_bf16<128,128,2,4,0,false,true,true><<<g, tblk, 0, stream>>>(
            Qp, Kp, nullptr, 0.125f, (void*)attn, DHEAD, EDIM, EDIM, SEQ,
            (long)SEQ * EDIM, (long)DHEAD,            // A: Qp + b*S*E + h*Dh
            (long)SEQ * EDIM, (long)DHEAD,            // B: Kp + b*S*E + h*Dh
            (long)NHEADS * SEQ * SEQ, (long)SEQ * SEQ // C: attn + z*S*S
        );
    }

    // 4) softmax rows in place
    softmax_rows_k<<<dim3(BATCH * NHEADS * SEQ), tblk, 0, stream>>>(attn);

    // 5) ctx = attn x V, per (b,h): M=SEQ, N=Dh, K=SEQ  (A = f32 attn, convert)
    {
        dim3 g(SEQ / 128, DHEAD / 64, BATCH * NHEADS);
        gemm_bf16<128,64,4,2,1,false,false,false><<<g, tblk, 0, stream>>>(
            attn, Vt, nullptr, 1.0f, (void*)ctx, SEQ, SEQ, SEQ, EDIM,
            (long)NHEADS * SEQ * SEQ, (long)SEQ * SEQ,       // A: attn + z*S*S
            (long)NHEADS * DHEAD * SEQ, (long)DHEAD * SEQ,   // B: Vt + z*Dh*S
            (long)SEQ * EDIM, (long)DHEAD                    // C: ctx + b*S*E + h*Dh
        );
    }

    // 6) out = ctx x Wo + bo  (A bf16 -> async)
    {
        dim3 g(TOKENS / 128, EDIM / 128, 1);
        gemm_bf16<128,128,2,4,0,true,false,true><<<g, tblk, 0, stream>>>(
            ctx, Wo_t, bo, 1.0f, (void*)out, EDIM, EDIM, EDIM, EDIM,
            0,0, 0,0, 0,0);
    }
}
